// SwinBlock_Light_58789512348125
// MI455X (gfx1250) — compile-verified, matrix operand order checked by reading, not run on hardware
//
#include <hip/hip_runtime.h>
#include <hip/hip_bf16.h>

typedef __attribute__((ext_vector_type(16))) _Float16 v16h;
typedef __attribute__((ext_vector_type(8)))  float    v8f;
typedef __attribute__((ext_vector_type(2)))  _Float16 h2v;

// ---------------------------------------------------------------------------
// WMMA helpers (CDNA5 v_wmma_f32_16x16x32_f16, wave32)
// ---------------------------------------------------------------------------
__device__ __forceinline__ v8f wmma_f16(v16h a, v16h b, v8f c) {
  // (neg_a, A, neg_b, B, c_mod, C, reuse_a, reuse_b)
  return __builtin_amdgcn_wmma_f32_16x16x32_f16(false, a, false, b, (short)0, c,
                                                false, false);
}

// K index (first of a packed f16 pair) for fragment register j, lane-half hi.
// 16-bit A 16x32 layout: V0..V3 -> K {0..7|8..15}+hi*8 ; V4..V7 -> +16.
__device__ __forceinline__ int frag_k(int j, int hi) {
  return (j < 4) ? (hi * 8 + 2 * j) : (16 + hi * 8 + 2 * (j - 4));
}

// Fragment where the lane-indexed dimension is a ROW of `base` (row-major, ld
// f16 elems/row).  A-matrix: row = m.  B-matrix when B[k][n] = base[n][k]
// (i.e. weight stored [out_ch][in_ch]): row = n.
__device__ __forceinline__ v16h frag_ld_row(const _Float16* base, int ld,
                                            int row0, int k0) {
  const int lane = threadIdx.x & 31;
  const int hi = lane >> 4;
  const _Float16* p = base + (size_t)(row0 + (lane & 15)) * ld + k0;
  v16h f;
#pragma unroll
  for (int j = 0; j < 8; ++j) {
    h2v t = *(const h2v*)(p + frag_k(j, hi));
    f[2 * j] = t.x;
    f[2 * j + 1] = t.y;
  }
  return f;
}

// B-fragment from row-major [K][N] storage: B[k][n] = base[k*ld + n].
__device__ __forceinline__ v16h frag_ld_kn(const _Float16* base, int ld,
                                           int k0, int n0) {
  const int lane = threadIdx.x & 31;
  const int hi = lane >> 4;
  const int n = n0 + (lane & 15);
  v16h f;
#pragma unroll
  for (int j = 0; j < 8; ++j) {
    int k = k0 + frag_k(j, hi);
    f[2 * j]     = base[(size_t)k * ld + n];
    f[2 * j + 1] = base[(size_t)(k + 1) * ld + n];
  }
  return f;
}

__device__ __forceinline__ float gelu_exact(float v) {
  return 0.5f * v * (1.0f + erff(v * 0.70710678118654752f));
}

#define Cdim 256
#define Ndim 16384   // H*W
#define HID  512

// ---------------------------------------------------------------------------
// Weight f32 -> f16 conversion (runs once; 1 MB result stays in L2)
// layout in ws: [0,196608) qkv_w | [196608,262144) proj_w |
//               [262144,393216) conv1_w | [393216,524288) conv3_w
// ---------------------------------------------------------------------------
__global__ __launch_bounds__(256) void k_cvt(const float* __restrict__ qkvw,
                                             const float* __restrict__ projw,
                                             const float* __restrict__ w1,
                                             const float* __restrict__ w3,
                                             _Float16* __restrict__ dst) {
  int idx = blockIdx.x * 256 + threadIdx.x;
  if (idx < 196608)       dst[idx] = (_Float16)qkvw[idx];
  else if (idx < 262144)  dst[idx] = (_Float16)projw[idx - 196608];
  else if (idx < 393216)  dst[idx] = (_Float16)w1[idx - 262144];
  else if (idx < 524288)  dst[idx] = (_Float16)w3[idx - 393216];
}

// ---------------------------------------------------------------------------
// Fused LN1 + QKV GEMM + chunk attention + proj + residual.
// One block (8 wave32s) per 64-token chunk; 2048 blocks.
// x is (B,C,H*W) f32; x2 out is (B,N,C) f32 token-major.
// ---------------------------------------------------------------------------
__global__ __launch_bounds__(256) void k_attn(
    const float* __restrict__ x, const float* __restrict__ ln1g,
    const float* __restrict__ ln1b, const _Float16* __restrict__ wqkv,
    const float* __restrict__ bqkv, const _Float16* __restrict__ wproj,
    const float* __restrict__ bproj, float* __restrict__ x2) {
  __shared__ _Float16 sA[64 * 258];  // LN-ed x, later attention output
  __shared__ _Float16 sQ[64 * 258];
  __shared__ _Float16 sK[64 * 258];
  __shared__ _Float16 sV[64 * 258];
  __shared__ float    sS[64 * 66];   // scores f32
  __shared__ _Float16 sP[64 * 66];   // softmax probs f16
  __shared__ float pm[256], pv[256], mu[64], rs[64];

  const int tid  = threadIdx.x;
  const int blk  = blockIdx.x;
  const int bimg = blk >> 8;            // image index (256 chunks / image)
  const int tok0 = (blk & 255) * 64;    // chunk start within N
  const float* xb = x + (size_t)bimg * Cdim * Ndim;

  // ---- LN1 pass 1: per-token mean/var (coalesced along tokens) ----
  const int t  = tid & 63;   // token within chunk
  const int cb = tid >> 6;   // channel quarter
  {
    float s = 0.f, ss = 0.f;
#pragma unroll 4
    for (int i = 0; i < 64; ++i) {
      int c = cb * 64 + i;
      float v = xb[(size_t)c * Ndim + tok0 + t];
      s += v; ss += v * v;
    }
    pm[tid] = s; pv[tid] = ss;
  }
  __syncthreads();
  if (tid < 64) {
    float s  = pm[tid] + pm[tid + 64] + pm[tid + 128] + pm[tid + 192];
    float ss = pv[tid] + pv[tid + 64] + pv[tid + 128] + pv[tid + 192];
    float m  = s * (1.0f / 256.0f);
    float vv = ss * (1.0f / 256.0f) - m * m;
    mu[tid] = m;
    rs[tid] = rsqrtf(vv + 1e-5f);
  }
  __syncthreads();
  // ---- LN1 pass 2: normalize -> f16 tile in LDS ----
  {
    float m = mu[t], r = rs[t];
#pragma unroll 4
    for (int i = 0; i < 64; ++i) {
      int c = cb * 64 + i;
      float v = xb[(size_t)c * Ndim + tok0 + t];
      sA[t * 258 + c] = (_Float16)((v - m) * r * ln1g[c] + ln1b[c]);
    }
  }
  __syncthreads();

  const int wv   = tid >> 5;
  const int lane = tid & 31;
  const int hi   = lane >> 4;
  const int ln   = lane & 15;

  // ---- QKV GEMM: (64x256) x (256x768) -> q,k,v tiles in LDS ----
  for (int nt = wv; nt < 48; nt += 8) {
    const int o0 = nt * 16;
    _Float16* dst = (o0 < 256) ? sQ : (o0 < 512) ? sK : sV;
    const int ch0  = o0 & 255;
    const float bb = bqkv[o0 + ln];
    for (int mt = 0; mt < 4; ++mt) {
      v8f acc = {};
#pragma unroll
      for (int ks = 0; ks < 8; ++ks) {
        v16h a = frag_ld_row(sA, 258, mt * 16, ks * 32);
        v16h b = frag_ld_row(wqkv, 256, o0, ks * 32);  // B[k][n] = W[o][c]
        acc = wmma_f16(a, b, acc);
      }
#pragma unroll
      for (int r = 0; r < 8; ++r)
        dst[(mt * 16 + r + 8 * hi) * 258 + ch0 + ln] = (_Float16)(acc[r] + bb);
    }
  }
  __syncthreads();

  // ---- scores = q @ k^T * C^-0.5 (16 tiles, 2 per wave) ----
#pragma unroll
  for (int i = 0; i < 2; ++i) {
    int tt = wv * 2 + i;
    int mt = tt >> 2, nt = tt & 3;
    v8f acc = {};
#pragma unroll
    for (int ks = 0; ks < 8; ++ks) {
      v16h a = frag_ld_row(sQ, 258, mt * 16, ks * 32);
      v16h b = frag_ld_row(sK, 258, nt * 16, ks * 32);  // B[k][n] = K[n][k]
      acc = wmma_f16(a, b, acc);
    }
#pragma unroll
    for (int r = 0; r < 8; ++r)
      sS[(mt * 16 + r + 8 * hi) * 66 + nt * 16 + ln] = acc[r] * 0.0625f;
  }
  __syncthreads();

  // ---- softmax over 64 keys per row (4 lanes / row, shfl reductions) ----
  {
    const int row = tid >> 2, seg = tid & 3;
    float vals[16];
    float mx = -3.4e38f;
#pragma unroll
    for (int i = 0; i < 16; ++i) {
      vals[i] = sS[row * 66 + seg * 16 + i];
      mx = fmaxf(mx, vals[i]);
    }
    mx = fmaxf(mx, __shfl_xor(mx, 1));
    mx = fmaxf(mx, __shfl_xor(mx, 2));
    float sum = 0.f;
#pragma unroll
    for (int i = 0; i < 16; ++i) { vals[i] = expf(vals[i] - mx); sum += vals[i]; }
    sum += __shfl_xor(sum, 1);
    sum += __shfl_xor(sum, 2);
    float inv = 1.0f / sum;
#pragma unroll
    for (int i = 0; i < 16; ++i)
      sP[row * 66 + seg * 16 + i] = (_Float16)(vals[i] * inv);
  }
  __syncthreads();

  // ---- attn @ v  -> reuse sA as (64x256) f16 output ----
#pragma unroll
  for (int i = 0; i < 8; ++i) {
    int tt = wv * 8 + i;
    int mt = tt >> 4, nt = tt & 15;
    v8f acc = {};
#pragma unroll
    for (int ks = 0; ks < 2; ++ks) {
      v16h a = frag_ld_row(sP, 66, mt * 16, ks * 32);
      v16h b = frag_ld_kn(sV, 258, ks * 32, nt * 16);  // B[k][n] = V[k][n]
      acc = wmma_f16(a, b, acc);
    }
#pragma unroll
    for (int r = 0; r < 8; ++r)
      sA[(mt * 16 + r + 8 * hi) * 258 + nt * 16 + ln] = (_Float16)acc[r];
  }
  __syncthreads();

  // ---- proj GEMM + bias + residual (original x), write x2 token-major ----
#pragma unroll
  for (int i = 0; i < 8; ++i) {
    int tt = wv * 8 + i;
    int mt = tt >> 4, nt = tt & 15;
    v8f acc = {};
#pragma unroll
    for (int ks = 0; ks < 8; ++ks) {
      v16h a = frag_ld_row(sA, 258, mt * 16, ks * 32);
      v16h b = frag_ld_row(wproj, 256, nt * 16, ks * 32);
      acc = wmma_f16(a, b, acc);
    }
    const int ch = nt * 16 + ln;
    const float pb = bproj[ch];
#pragma unroll
    for (int r = 0; r < 8; ++r) {
      int m = mt * 16 + r + 8 * hi;
      int tok = tok0 + m;
      float res = xb[(size_t)ch * Ndim + tok];
      x2[((size_t)bimg * Ndim + tok) * Cdim + ch] = acc[r] + pb + res;
    }
  }
}

// ---------------------------------------------------------------------------
// Fused LN2 + conv1 (1x1, 256->512) + exact GELU.  64 tokens per block.
// x2 is (B*N, C) f32; h1 out is (B*N, 512) f16.
// ---------------------------------------------------------------------------
__global__ __launch_bounds__(256) void k_ffn1(
    const float* __restrict__ x2, const float* __restrict__ ln2g,
    const float* __restrict__ ln2b, const _Float16* __restrict__ w1,
    const float* __restrict__ b1, _Float16* __restrict__ h1) {
  __shared__ _Float16 sA[64 * 258];
  const int tid = threadIdx.x;
  const size_t tok0 = (size_t)blockIdx.x * 64;

  // LN2: 4 lanes per token, shfl reduction within the quad
  {
    const int t = tid >> 2, seg = tid & 3;
    const float* row = x2 + (tok0 + t) * Cdim;
    float s = 0.f, ss = 0.f;
#pragma unroll 4
    for (int i = 0; i < 64; ++i) {
      float v = row[seg * 64 + i];
      s += v; ss += v * v;
    }
    s  += __shfl_xor(s, 1);  s  += __shfl_xor(s, 2);
    ss += __shfl_xor(ss, 1); ss += __shfl_xor(ss, 2);
    float m  = s * (1.0f / 256.0f);
    float vv = ss * (1.0f / 256.0f) - m * m;
    float r  = rsqrtf(vv + 1e-5f);
#pragma unroll 4
    for (int i = 0; i < 64; ++i) {
      int c = seg * 64 + i;
      sA[t * 258 + c] = (_Float16)((row[c] - m) * r * ln2g[c] + ln2b[c]);
    }
  }
  __syncthreads();

  const int wv = tid >> 5, lane = tid & 31, hi = lane >> 4, ln = lane & 15;
  for (int nt = wv; nt < 32; nt += 8) {
    const int hc0 = nt * 16;
    const float bb = b1[hc0 + ln];
    for (int mt = 0; mt < 4; ++mt) {
      v8f acc = {};
#pragma unroll
      for (int ks = 0; ks < 8; ++ks) {
        v16h a = frag_ld_row(sA, 258, mt * 16, ks * 32);
        v16h b = frag_ld_row(w1, 256, hc0, ks * 32);
        acc = wmma_f16(a, b, acc);
      }
#pragma unroll
      for (int r = 0; r < 8; ++r) {
        float v = gelu_exact(acc[r] + bb);
        h1[(tok0 + mt * 16 + r + 8 * hi) * HID + hc0 + ln] = (_Float16)v;
      }
    }
  }
}

// ---------------------------------------------------------------------------
// Depthwise 3x3 conv (pad 1) + exact GELU.  One block per spatial position,
// each thread handles a packed f16 channel pair.
// ---------------------------------------------------------------------------
__global__ __launch_bounds__(256) void k_dwconv(
    const _Float16* __restrict__ h1, const float* __restrict__ w2,
    const float* __restrict__ b2, _Float16* __restrict__ h2) {
  const int tid  = threadIdx.x;
  const int blk  = blockIdx.x;          // b * 16384 + pos
  const int bimg = blk >> 14;
  const int pos  = blk & 16383;
  const int y = pos >> 7, xq = pos & 127;
  const int c0 = tid * 2;
  float a0 = 0.f, a1 = 0.f;
#pragma unroll
  for (int dy = -1; dy <= 1; ++dy) {
    int yy = y + dy;
    if (yy < 0 || yy > 127) continue;
#pragma unroll
    for (int dx = -1; dx <= 1; ++dx) {
      int xx = xq + dx;
      if (xx < 0 || xx > 127) continue;
      h2v p = *(const h2v*)(h1 + ((size_t)bimg * 16384 + yy * 128 + xx) * HID + c0);
      int wi = (dy + 1) * 3 + (dx + 1);
      a0 += (float)p.x * w2[c0 * 9 + wi];
      a1 += (float)p.y * w2[(c0 + 1) * 9 + wi];
    }
  }
  a0 = gelu_exact(a0 + b2[c0]);
  a1 = gelu_exact(a1 + b2[c0 + 1]);
  h2v o; o.x = (_Float16)a0; o.y = (_Float16)a1;
  *(h2v*)(h2 + (size_t)blk * HID + c0) = o;
}

// ---------------------------------------------------------------------------
// conv3 (1x1, 512->256) + bias + residual(x2) with LDS transpose epilogue so
// the (B,C,H,W) output is written as contiguous token runs.
// Grid: (M/64) * (C/64) blocks.
// ---------------------------------------------------------------------------
__global__ __launch_bounds__(256) void k_ffn2(
    const _Float16* __restrict__ h2, const _Float16* __restrict__ w3,
    const float* __restrict__ b3, const float* __restrict__ x2,
    float* __restrict__ out) {
  __shared__ float ot[64 * 65];  // [ch_local][token_local]
  const int tid = threadIdx.x;
  const int blk = blockIdx.x;
  const size_t m0 = (size_t)(blk >> 2) * 64;   // global token base
  const int ch0 = (blk & 3) * 64;
  const int wv = tid >> 5, lane = tid & 31, hi = lane >> 4, ln = lane & 15;

#pragma unroll
  for (int i = 0; i < 2; ++i) {
    int tt = wv * 2 + i;
    int mt = tt >> 2, nt = tt & 3;
    v8f acc = {};
#pragma unroll
    for (int ks = 0; ks < 16; ++ks) {
      v16h a = frag_ld_row(h2 + m0 * HID, HID, mt * 16, ks * 32);
      v16h b = frag_ld_row(w3, HID, ch0 + nt * 16, ks * 32);
      acc = wmma_f16(a, b, acc);
    }
    const int ch = ch0 + nt * 16 + ln;
    const float bb = b3[ch];
#pragma unroll
    for (int r = 0; r < 8; ++r) {
      int ml = mt * 16 + r + 8 * hi;
      float v = acc[r] + bb + x2[(m0 + ml) * Cdim + ch];
      ot[(nt * 16 + ln) * 65 + ml] = v;
    }
  }
  __syncthreads();

  // transpose write-out: contiguous 16-float runs along tokens per channel
  const int chl = tid >> 2, seg = tid & 3;
  const int bimg = (int)(m0 >> 14);
  const int nimg = (int)(m0 & 16383);
  float* op = out + ((size_t)bimg * Cdim + ch0 + chl) * Ndim + nimg + seg * 16;
#pragma unroll
  for (int i = 0; i < 16; ++i) op[i] = ot[chl * 65 + seg * 16 + i];
}

// ---------------------------------------------------------------------------
extern "C" void kernel_launch(void* const* d_in, const int* in_sizes, int n_in,
                              void* d_out, int out_size, void* d_ws,
                              size_t ws_size, hipStream_t stream) {
  const float* x     = (const float*)d_in[0];
  const float* ln1g  = (const float*)d_in[1];
  const float* ln1b  = (const float*)d_in[2];
  const float* qkvw  = (const float*)d_in[3];
  const float* qkvb  = (const float*)d_in[4];
  const float* projw = (const float*)d_in[5];
  const float* projb = (const float*)d_in[6];
  const float* ln2g  = (const float*)d_in[7];
  const float* ln2b  = (const float*)d_in[8];
  const float* w1    = (const float*)d_in[9];
  const float* b1    = (const float*)d_in[10];
  const float* w2    = (const float*)d_in[11];
  const float* b2    = (const float*)d_in[12];
  const float* w3    = (const float*)d_in[13];
  const float* b3    = (const float*)d_in[14];

  char* ws = (char*)d_ws;
  _Float16* wh    = (_Float16*)ws;          // 524288 f16 = 1 MB
  _Float16* wqkvh = wh;
  _Float16* wprojh= wh + 196608;
  _Float16* w1h   = wh + 262144;
  _Float16* w3h   = wh + 393216;
  const size_t MB = 1u << 20;
  float*    x2 = (float*)(ws + MB);                            // 134 MB f32
  _Float16* h1 = (_Float16*)(ws + MB + 134217728ull);          // 134 MB f16
  _Float16* h2 = (_Float16*)(ws + MB + 2ull * 134217728ull);   // 134 MB f16

  k_cvt<<<2048, 256, 0, stream>>>(qkvw, projw, w1, w3, wh);
  k_attn<<<2048, 256, 0, stream>>>(x, ln1g, ln1b, wqkvh, qkvb, wprojh, projb, x2);
  k_ffn1<<<2048, 256, 0, stream>>>(x2, ln2g, ln2b, w1h, b1, h1);
  k_dwconv<<<131072, 256, 0, stream>>>(h1, w2, b2, h2);
  k_ffn2<<<8192, 256, 0, stream>>>(h2, w3h, b3, x2, (float*)d_out);
}
